// EdgeDecoder_26061861552143
// MI455X (gfx1250) — matrix operand order repaired and checked by reading, not compiled, not run on hardware
//
#include <hip/hip_runtime.h>

typedef __attribute__((ext_vector_type(16))) _Float16 v16h;
typedef __attribute__((ext_vector_type(8)))  _Float16 v8h;
typedef __attribute__((ext_vector_type(8)))  float    v8f;
typedef __attribute__((ext_vector_type(4)))  float    v4f;
typedef __attribute__((ext_vector_type(4)))  int      v4i;

#define LDS_FENCE() asm volatile("s_wait_dscnt 0" ::: "memory")

static constexpr int D   = 128;   // node feature dim
static constexpr int D2  = 256;   // feats / hidden1 width
static constexpr int H2  = 128;   // hidden2 width
static constexpr int WAVES = 4;
static constexpr int BLOCK = WAVES * 32;

static constexpr int W1F_ELEMS = D2 * D2;   // 65536 f16 fragments (8 ktile x 16 ntile)
static constexpr int W2F_ELEMS = D2 * H2;   // 32768 f16 fragments (8 ktile x 8 ntile)
static constexpr size_t SM_WEIGHTS  = (size_t)(W1F_ELEMS + W2F_ELEMS) * sizeof(_Float16); // 196608
static constexpr size_t SM_PER_WAVE = (size_t)16 * D2 * sizeof(float)      // yBuf f32 [16][256]
                                    + (size_t)16 * D2 * sizeof(_Float16);  // act  f16 [16][256]
static constexpr int PARAM_FLOATS = 256 * 5 + 128 * 5;                     // 1920
static constexpr size_t SMEM_BYTES = SM_WEIGHTS + WAVES * SM_PER_WAVE
                                   + (size_t)PARAM_FLOATS * sizeof(float); // ~295 KB <= 320 KB
static_assert(SMEM_BYTES <= 320u * 1024u, "LDS budget exceeded");

// ---------------------------------------------------------------------------
// Pre-swizzle w1 (f32 [256,256]) and w2 (f32 [256,128]) into WMMA B-fragment
// f16 layout: frag(kt,nt) -> 512 f16, lane L holds n = nt*16 + (L%16),
// k = kt*32 + (L>=16 ? 16 : 0) + j  for j = 0..15 contiguous per lane.
// ---------------------------------------------------------------------------
__global__ void prep_weights_kernel(const float* __restrict__ w1,
                                    const float* __restrict__ w2,
                                    _Float16* __restrict__ wf) {
    int idx = blockIdx.x * blockDim.x + threadIdx.x;
    if (idx < W1F_ELEMS) {
        int frag = idx >> 9, r = idx & 511;
        int lane = r >> 4, j = r & 15;
        int kt = frag >> 4, nt = frag & 15;
        int k = kt * 32 + ((lane >> 4) << 4) + j;
        int n = nt * 16 + (lane & 15);
        wf[idx] = (_Float16)w1[k * D2 + n];
    } else if (idx < W1F_ELEMS + W2F_ELEMS) {
        int t = idx - W1F_ELEMS;
        int frag = t >> 9, r = t & 511;
        int lane = r >> 4, j = r & 15;
        int kt = frag >> 3, nt = frag & 7;
        int k = kt * 32 + ((lane >> 4) << 4) + j;
        int n = nt * 16 + (lane & 15);
        wf[idx] = (_Float16)w2[k * H2 + n];
    }
}

// ---------------------------------------------------------------------------
// Main fused edge-MLP kernel: one wave per 16-edge tile.
// ---------------------------------------------------------------------------
__global__ __launch_bounds__(BLOCK) void edge_mlp_kernel(
    const float* __restrict__ x,
    const int*   __restrict__ ei,
    const float* __restrict__ g0,  const float* __restrict__ bb0,
    const float* __restrict__ b1,  const float* __restrict__ g1, const float* __restrict__ bb1,
    const float* __restrict__ b2,  const float* __restrict__ g2, const float* __restrict__ bb2,
    const float* __restrict__ wp,  const float* __restrict__ bp,
    const float* __restrict__ ww,  const float* __restrict__ bw,
    const _Float16* __restrict__ wf,
    float* __restrict__ out,
    int E, int tileCount)
{
    extern __shared__ __align__(16) char smem[];
    _Float16* w1f = (_Float16*)smem;
    _Float16* w2f = w1f + W1F_ELEMS;
    const int wave = threadIdx.x >> 5;
    const int lane = threadIdx.x & 31;
    float*    yBuf = (float*)(smem + SM_WEIGHTS + (size_t)wave * SM_PER_WAVE);
    _Float16* act  = (_Float16*)(yBuf + 16 * D2);
    float* prm  = (float*)(smem + SM_WEIGHTS + (size_t)WAVES * SM_PER_WAVE);
    float* sg0  = prm;          // 256
    float* sb0  = sg0  + 256;   // 256
    float* sb1  = sb0  + 256;   // linear1 bias, 256
    float* sg1  = sb1  + 256;   // 256
    float* sbb1 = sg1  + 256;   // 256
    float* sb2  = sbb1 + 256;   // linear2 bias, 128
    float* sg2  = sb2  + 128;   // 128
    float* sbb2 = sg2  + 128;   // 128
    float* swp  = sbb2 + 128;   // 128
    float* sww  = swp  + 128;   // 128

    // ---- stage swizzled weights into LDS (once per block) ----
#if __has_builtin(__builtin_amdgcn_global_load_async_to_lds_b128)
    {
        // CDNA5 async global->LDS DMA path (ASYNCcnt-tracked, no VGPR round trip).
        // Builtin signature (from clang diagnostic): (AS1 int4*, AS3 int4*, imm, imm).
        typedef __attribute__((address_space(1))) v4i* gv4ip;
        typedef __attribute__((address_space(3))) v4i* lv4ip;
        gv4ip gsrc = (gv4ip)(const void*)wf;
        lv4ip ldst = (lv4ip)smem;
        const int n16 = (int)(SM_WEIGHTS / 16);
        for (int i = threadIdx.x; i < n16; i += BLOCK)
            __builtin_amdgcn_global_load_async_to_lds_b128(gsrc + i, ldst + i, 0, 0);
        asm volatile("s_wait_asynccnt 0" ::: "memory");
    }
#else
    {
        const uint4* src = (const uint4*)wf;
        uint4* dst = (uint4*)w1f;
        const int n16 = (int)(SM_WEIGHTS / 16);
        for (int i = threadIdx.x; i < n16; i += BLOCK) dst[i] = src[i];
    }
#endif
    // ---- stage LN / bias / head params into LDS ----
    for (int i = threadIdx.x; i < 256; i += BLOCK) {
        sg0[i] = g0[i]; sb0[i] = bb0[i]; sb1[i] = b1[i];
        sg1[i] = g1[i]; sbb1[i] = bb1[i];
    }
    for (int i = threadIdx.x; i < 128; i += BLOCK) {
        sb2[i] = b2[i]; sg2[i] = g2[i]; sbb2[i] = bb2[i];
        swp[i] = wp[i]; sww[i] = ww[i];
    }
    __syncthreads();

    const float bpv = bp[0], bwv = bw[0];
    const int e2   = lane >> 1, part = lane & 1;  // 2 lanes per edge for LN/heads
    const int mCD  = lane & 15, hiCD = lane >> 4; // WMMA C/D layout: n=mCD, rows hiCD*8+r
    const int kbA  = (lane >> 4) * 16;            // WMMA A-frag K sub-base: contiguous k = 16h + s, as in B prep

    for (int tile = blockIdx.x * WAVES + wave; tile < tileCount; tile += gridDim.x * WAVES) {
        // ---------- gather + feats + LN0 stats (fused) ----------
        const int eg  = tile * 16 + e2;
        const int egc = (eg < E) ? eg : 0;
        const int sN = ei[egc], dN = ei[E + egc];
        const v4f* xs = (const v4f*)(x + (size_t)sN * D + part * 64);
        const v4f* xd = (const v4f*)(x + (size_t)dN * D + part * 64);
        float* yrow = yBuf + e2 * D2;
        v4f* yadd = (v4f*)(yrow + part * 64);
        v4f* ymul = (v4f*)(yrow + D + part * 64);
        float sum = 0.f, sq = 0.f;
        #pragma unroll
        for (int i = 0; i < 16; ++i) {
            v4f a = xs[i], b = xd[i];
            v4f s0 = a + b;
            v4f m0 = a * b;
            yadd[i] = s0; ymul[i] = m0;
            sum += s0[0]+s0[1]+s0[2]+s0[3] + m0[0]+m0[1]+m0[2]+m0[3];
            sq  += s0[0]*s0[0]+s0[1]*s0[1]+s0[2]*s0[2]+s0[3]*s0[3]
                 + m0[0]*m0[0]+m0[1]*m0[1]+m0[2]*m0[2]+m0[3]*m0[3];
        }
        sum += __shfl_xor(sum, 1, 32);
        sq  += __shfl_xor(sq, 1, 32);
        float mean = sum * (1.f / 256.f);
        float rstd = rsqrtf(sq * (1.f / 256.f) - mean * mean + 1e-5f);
        LDS_FENCE();
        // ---------- LN0 normalize -> act f16 [16][256] ----------
        {
            const float* yr = yBuf + e2 * D2 + part * 128;
            _Float16* ar = act + e2 * D2 + part * 128;
            #pragma unroll
            for (int kk = 0; kk < 128; kk += 8) {
                v4f v0 = *(const v4f*)(yr + kk);
                v4f v1 = *(const v4f*)(yr + kk + 4);
                const int k = part * 128 + kk;
                v8h o;
                #pragma unroll
                for (int c = 0; c < 4; ++c) {
                    o[c]     = (_Float16)((v0[c] - mean) * rstd * sg0[k + c]     + sb0[k + c]);
                    o[c + 4] = (_Float16)((v1[c] - mean) * rstd * sg0[k + 4 + c] + sb0[k + 4 + c]);
                }
                *(v8h*)(ar + kk) = o;
            }
        }
        LDS_FENCE();
        // ---------- GEMM1: h1 = act[16,256] @ w1[256,256] (WMMA f16 -> f32) ----------
        {
            // hoist all 8 A fragments (invariant across nt) into registers
            v16h afrag[8];
            #pragma unroll
            for (int kt = 0; kt < 8; ++kt) {
                const _Float16* ar = act + mCD * D2 + kt * 32;
                v8h alo = *(const v8h*)(ar + kbA);
                v8h ahi = *(const v8h*)(ar + kbA + 8);
                #pragma unroll
                for (int i = 0; i < 8; ++i) { afrag[kt][i] = alo[i]; afrag[kt][8 + i] = ahi[i]; }
            }
            for (int nt = 0; nt < 16; ++nt) {
                v8f acc = {};
                #pragma unroll
                for (int kt = 0; kt < 8; ++kt) {
                    v16h b = *(const v16h*)(w1f + (size_t)(kt * 16 + nt) * 512 + lane * 16);
                    acc = __builtin_amdgcn_wmma_f32_16x16x32_f16(false, afrag[kt], false, b,
                                                                 (short)0, acc, false, false);
                }
                const int n = nt * 16 + mCD;
                const float bias = sb1[n];
                #pragma unroll
                for (int r = 0; r < 8; ++r)
                    yBuf[(hiCD * 8 + r) * D2 + n] = acc[r] + bias;
            }
        }
        LDS_FENCE();
        // ---------- LN1 + ReLU -> act f16 [16][256] ----------
        {
            const float* yrf = yBuf + e2 * D2 + part * 128;
            float s1 = 0.f, q1 = 0.f;
            #pragma unroll
            for (int i = 0; i < 32; ++i) {
                v4f v = ((const v4f*)yrf)[i];
                s1 += v[0]+v[1]+v[2]+v[3];
                q1 += v[0]*v[0]+v[1]*v[1]+v[2]*v[2]+v[3]*v[3];
            }
            s1 += __shfl_xor(s1, 1, 32);
            q1 += __shfl_xor(q1, 1, 32);
            float mu = s1 * (1.f / 256.f);
            float rs = rsqrtf(q1 * (1.f / 256.f) - mu * mu + 1e-5f);
            _Float16* ar = act + e2 * D2 + part * 128;
            #pragma unroll
            for (int kk = 0; kk < 128; kk += 8) {
                v4f v0 = *(const v4f*)(yrf + kk);
                v4f v1 = *(const v4f*)(yrf + kk + 4);
                const int k = part * 128 + kk;
                v8h o;
                #pragma unroll
                for (int c = 0; c < 4; ++c) {
                    float t0 = (v0[c] - mu) * rs * sg1[k + c]     + sbb1[k + c];
                    float t1 = (v1[c] - mu) * rs * sg1[k + 4 + c] + sbb1[k + 4 + c];
                    o[c]     = (_Float16)(t0 > 0.f ? t0 : 0.f);
                    o[c + 4] = (_Float16)(t1 > 0.f ? t1 : 0.f);
                }
                *(v8h*)(ar + kk) = o;
            }
        }
        LDS_FENCE();
        // ---------- GEMM2: h2 = act[16,256] @ w2[256,128] ----------
        {
            v16h afrag[8];
            #pragma unroll
            for (int kt = 0; kt < 8; ++kt) {
                const _Float16* ar = act + mCD * D2 + kt * 32;
                v8h alo = *(const v8h*)(ar + kbA);
                v8h ahi = *(const v8h*)(ar + kbA + 8);
                #pragma unroll
                for (int i = 0; i < 8; ++i) { afrag[kt][i] = alo[i]; afrag[kt][8 + i] = ahi[i]; }
            }
            for (int nt = 0; nt < 8; ++nt) {
                v8f acc = {};
                #pragma unroll
                for (int kt = 0; kt < 8; ++kt) {
                    v16h b = *(const v16h*)(w2f + (size_t)(kt * 8 + nt) * 512 + lane * 16);
                    acc = __builtin_amdgcn_wmma_f32_16x16x32_f16(false, afrag[kt], false, b,
                                                                 (short)0, acc, false, false);
                }
                const int n = nt * 16 + mCD;
                const float bias = sb2[n];
                #pragma unroll
                for (int r = 0; r < 8; ++r)
                    yBuf[(hiCD * 8 + r) * H2 + n] = acc[r] + bias;
            }
        }
        LDS_FENCE();
        // ---------- LN2 + ReLU -> act f16 [16][128] ----------
        {
            const float* yrf = yBuf + e2 * H2 + part * 64;
            float s2 = 0.f, q2 = 0.f;
            #pragma unroll
            for (int i = 0; i < 16; ++i) {
                v4f v = ((const v4f*)yrf)[i];
                s2 += v[0]+v[1]+v[2]+v[3];
                q2 += v[0]*v[0]+v[1]*v[1]+v[2]*v[2]+v[3]*v[3];
            }
            s2 += __shfl_xor(s2, 1, 32);
            q2 += __shfl_xor(q2, 1, 32);
            float mu = s2 * (1.f / 128.f);
            float rs = rsqrtf(q2 * (1.f / 128.f) - mu * mu + 1e-5f);
            _Float16* ar = act + e2 * H2 + part * 64;
            #pragma unroll
            for (int kk = 0; kk < 64; kk += 8) {
                v4f v0 = *(const v4f*)(yrf + kk);
                v4f v1 = *(const v4f*)(yrf + kk + 4);
                const int k = part * 64 + kk;
                v8h o;
                #pragma unroll
                for (int c = 0; c < 4; ++c) {
                    float t0 = (v0[c] - mu) * rs * sg2[k + c]     + sbb2[k + c];
                    float t1 = (v1[c] - mu) * rs * sg2[k + 4 + c] + sbb2[k + 4 + c];
                    o[c]     = (_Float16)(t0 > 0.f ? t0 : 0.f);
                    o[c + 4] = (_Float16)(t1 > 0.f ? t1 : 0.f);
                }
                *(v8h*)(ar + kk) = o;
            }
        }
        LDS_FENCE();
        // ---------- heads: prob = h@wp + bp ; weight = relu(h@ww + bw) ----------
        {
            const _Float16* hr = act + e2 * H2 + part * 64;
            const float* wpp = swp + part * 64;
            const float* wwp = sww + part * 64;
            float p = 0.f, w = 0.f;
            #pragma unroll
            for (int kk = 0; kk < 64; kk += 8) {
                v8h h8 = *(const v8h*)(hr + kk);
                #pragma unroll
                for (int c = 0; c < 8; ++c) {
                    float h = (float)h8[c];
                    p += h * wpp[kk + c];
                    w += h * wwp[kk + c];
                }
            }
            p += __shfl_xor(p, 1, 32);
            w += __shfl_xor(w, 1, 32);
            if (part == 0 && eg < E) {
                out[eg] = p + bpv;
                float t = w + bwv;
                out[E + eg] = t > 0.f ? t : 0.f;
            }
        }
        // prefetch next tile's edge indices into cache
        {
            int ntile = tile + gridDim.x * WAVES;
            if (ntile < tileCount) __builtin_prefetch(ei + ntile * 16, 0, 1);
        }
        LDS_FENCE();
    }
}

extern "C" void kernel_launch(void* const* d_in, const int* in_sizes, int n_in,
                              void* d_out, int out_size, void* d_ws, size_t ws_size,
                              hipStream_t stream) {
    const float* x   = (const float*)d_in[0];
    const int*   ei  = (const int*)d_in[1];
    const float* g0  = (const float*)d_in[2];
    const float* b0  = (const float*)d_in[3];
    const float* w1  = (const float*)d_in[4];
    const float* b1  = (const float*)d_in[5];
    const float* g1  = (const float*)d_in[6];
    const float* bb1 = (const float*)d_in[7];
    const float* w2  = (const float*)d_in[8];
    const float* b2  = (const float*)d_in[9];
    const float* g2  = (const float*)d_in[10];
    const float* bb2 = (const float*)d_in[11];
    const float* wp  = (const float*)d_in[12];
    const float* bp  = (const float*)d_in[13];
    const float* ww  = (const float*)d_in[14];
    const float* bw  = (const float*)d_in[15];
    float* out = (float*)d_out;

    const int E = in_sizes[1] / 2;
    const int tileCount = (E + 15) / 16;
    _Float16* wf = (_Float16*)d_ws;

    prep_weights_kernel<<<(W1F_ELEMS + W2F_ELEMS + 255) / 256, 256, 0, stream>>>(w1, w2, wf);

    (void)hipFuncSetAttribute((const void*)edge_mlp_kernel,
                              hipFuncAttributeMaxDynamicSharedMemorySize,
                              (int)SMEM_BYTES);
    int grid = 1024;
    const int maxGrid = (tileCount + WAVES - 1) / WAVES;
    if (grid > maxGrid) grid = maxGrid;
    if (grid < 1) grid = 1;
    edge_mlp_kernel<<<grid, BLOCK, SMEM_BYTES, stream>>>(
        x, ei, g0, b0, b1, g1, bb1, b2, g2, bb2, wp, bp, ww, bw, wf, out, E, tileCount);
}
